// SupConBinaryLoss_25220047962606
// MI455X (gfx1250) — matrix operand order repaired and checked by reading, not compiled, not run on hardware
//
#include <hip/hip_runtime.h>
#include <hip/hip_bf16.h>

// ---------------------------------------------------------------------------
// SupCon binary loss, fused for gfx1250 (MI455X):
//   sim = z z^T via v_wmma_f32_16x16x32_bf16 (f32 accum, dual accumulators to
//   break the WMMA RAW chain), geodesic epilogue (clip/acos/scale) + online
//   logsumexp + top-K negative mining fused in registers, per-row merge
//   through 320KB-class LDS, deterministic finalize.
// ---------------------------------------------------------------------------

typedef __attribute__((ext_vector_type(16))) __bf16 v16bf;
typedef __attribute__((ext_vector_type(8)))  __bf16 v8bf;
typedef __attribute__((ext_vector_type(8)))  float  v8f;
typedef __attribute__((ext_vector_type(4)))  unsigned int v4u;

#define B_ROWS 8192
#define DDIM   256
#define NTILE  (B_ROWS / 16)          // 512 column tiles
#define NWAVE  8                      // waves per workgroup (wave32)
#define INV_TAU 5.0f                  // 1 / 0.2
#define TWO_OVER_PI 0.636619772367581343f
#define NEGINF (-__builtin_inff())

// A-tile LDS row stride (ushorts): 256 + 8 pad -> 528B rows, 16B aligned,
// rotates rows across LDS banks so ds b128 A-fragment loads are conflict-free.
#define A_STRIDE 264
// dynamic LDS layout (bytes):
//   [0,8192)        labels  (u8, all B rows)
//   [8192,16640)    A tile  (16 x A_STRIDE bf16-bits)
//   [16640,82176)   reduction scratch (16384 floats)
#define SMEM_BYTES 82176

__device__ __forceinline__ unsigned short f2bf(float f) {
  unsigned u = __float_as_uint(f);
  u += 0x7FFFu + ((u >> 16) & 1u);   // round-to-nearest-even
  return (unsigned short)(u >> 16);
}

__device__ __forceinline__ void lse_add(float& m, float& s, float x) {
  if (x > m) { s = s * __expf(m - x) + 1.0f; m = x; }
  else       { s += __expf(x - m); }
}

__device__ __forceinline__ void lse_merge(float& m, float& s, float m2, float s2) {
  if (!(s2 > 0.0f)) return;          // empty partial (m2 == -inf)
  if (m2 > m) { s = s * __expf(m - m2) + s2; m = m2; }
  else        { s += s2 * __expf(m2 - m); }
}

// ---------------------------------------------------------------- kernel 0 --
__global__ void __launch_bounds__(256)
supcon_cvt(const float* __restrict__ z, unsigned short* __restrict__ zb) {
  int i = (blockIdx.x * blockDim.x + threadIdx.x) * 4;
  #pragma unroll
  for (int j = 0; j < 4; ++j) zb[i + j] = f2bf(z[i + j]);
}

// ---------------------------------------------------------------- kernel 1 --
__global__ void __launch_bounds__(256)
supcon_main(const unsigned short* __restrict__ zb,
            const int* __restrict__ labels,
            const int* __restrict__ topk,
            float* __restrict__ lossF, float* __restrict__ valF,
            float* __restrict__ lossM, float* __restrict__ valM) {
  extern __shared__ char smem[];
  unsigned char*  lab = (unsigned char*)smem;                 // 8192 u8
  unsigned short* At  = (unsigned short*)(smem + 8192);       // 16 x A_STRIDE
  float* red = (float*)(smem + 16640);                        // 16384 floats
  float* MALL = red;              // [16][128]
  float* SALL = red + 2048;
  float* MPOS = red + 4096;
  float* SPOS = red + 6144;
  float* SUMP = red + 8192;
  float* CNTP = red + 10240;
  float* CAND = red + 12288;      // [16][256]

  const int tid   = threadIdx.x;
  const int wave  = tid >> 5;
  const int lane  = tid & 31;
  const int col16 = lane & 15;
  const int half  = lane >> 4;
  const int r0    = blockIdx.x * 16;

  // Stage labels (all rows) and this block's A tile (bf16 bits) into LDS.
  for (int i = tid; i < B_ROWS; i += 256) lab[i] = (unsigned char)labels[i];
  {
    const int row = tid >> 4, seg = tid & 15;
    const v4u* src = (const v4u*)(zb + (size_t)(r0 + row) * DDIM + seg * 16);
    v4u* dst = (v4u*)(At + row * A_STRIDE + seg * 16);
    dst[0] = src[0];
    dst[1] = src[1];
  }
  __syncthreads();

  // Row labels for this lane's 8 accumulator rows (m = half*8 + v).
  int rl[8];
  #pragma unroll
  for (int v = 0; v < 8; ++v) rl[v] = lab[r0 + half * 8 + v];

  // A fragments (16x32 bf16 per k-step), resident in VGPRs across all tiles.
  // Layout: lanes 0-15 row=lane, K k0..k0+15;
  //         lanes 16-31 row=lane-16, K k0+16..k0+31.
  v16bf afrag[8];
  {
    const unsigned short* Arow = At + (size_t)col16 * A_STRIDE;
    const int klo = half ? 16 : 0;
    #pragma unroll
    for (int kk = 0; kk < 8; ++kk) {
      v8bf lo = *(const v8bf*)(Arow + kk * 32 + klo);
      v8bf hi = *(const v8bf*)(Arow + kk * 32 + klo + 8);
      afrag[kk] = __builtin_shufflevector(lo, hi,
          0, 1, 2, 3, 4, 5, 6, 7, 8, 9, 10, 11, 12, 13, 14, 15);
    }
  }

  // Per-lane fused accumulators for 8 rows each.
  float mAll[8], sAll[8], mP[8], sP[8], suP[8], cnP[8], t1[8], t2[8];
  #pragma unroll
  for (int v = 0; v < 8; ++v) {
    mAll[v] = NEGINF; sAll[v] = 0.f;
    mP[v]   = NEGINF; sP[v]   = 0.f;
    suP[v]  = 0.f;    cnP[v]  = 0.f;
    t1[v]   = NEGINF; t2[v]   = NEGINF;
  }

  // Each wave streams 64 of the 512 column tiles.
  for (int j = wave; j < NTILE; j += NWAVE) {
    const unsigned short* Brow =
        zb + (size_t)(j * 16 + col16) * DDIM + half * 16;
    // Branchless prefetch of the next tile's B rows (clamped; a redundant
    // prefetch of a resident line is harmless).
    {
      const int jn = (j + NWAVE < NTILE) ? (j + NWAVE) : j;
      __builtin_prefetch(zb + (size_t)(jn * 16 + col16) * DDIM + half * 16,
                         0, 3);
    }

    // Two accumulators: breaks the WMMA->WMMA RAW chain on C so the matrix
    // pipe runs at issue rate instead of instruction latency.
    v8f c0 = {0.f, 0.f, 0.f, 0.f, 0.f, 0.f, 0.f, 0.f};
    v8f c1 = {0.f, 0.f, 0.f, 0.f, 0.f, 0.f, 0.f, 0.f};
    #pragma unroll
    for (int kk = 0; kk < 8; kk += 2) {
      // B 32x16: lanes 0-15 col=lane K=k0..k0+15; lanes 16-31 K=k0+16..k0+31
      v16bf b0 = *(const v16bf*)(Brow + kk * 32);
      v16bf b1 = *(const v16bf*)(Brow + (kk + 1) * 32);
      c0 = __builtin_amdgcn_wmma_f32_16x16x32_bf16(
               false, afrag[kk], false, b0, (short)0, c0, false, false);
      c1 = __builtin_amdgcn_wmma_f32_16x16x32_bf16(
               false, afrag[kk + 1], false, b1, (short)0, c1, false, false);
    }
    v8f c = c0 + c1;

    const int nglob = j * 16 + col16;
    const int cl = lab[nglob];
    #pragma unroll
    for (int v = 0; v < 8; ++v) {
      const int mglob = r0 + half * 8 + v;
      if (nglob == mglob) continue;  // diagonal -> -inf, contributes nothing
      float d = c[v];
      d = fminf(fmaxf(d, -1.0f + 1e-7f), 1.0f - 1e-7f);
      float logit = (1.0f - TWO_OVER_PI * acosf(d)) * INV_TAU;
      lse_add(mAll[v], sAll[v], logit);
      if (cl == rl[v]) {
        suP[v] += logit;
        cnP[v] += 1.0f;
        lse_add(mP[v], sP[v], logit);
      } else {
        if (logit > t1[v]) { t2[v] = t1[v]; t1[v] = logit; }
        else if (logit > t2[v]) { t2[v] = logit; }
      }
    }
  }

  // Dump per-(wave,lane) partials; row m gets 128 partials + 256 candidates.
  {
    const int p = wave * 16 + col16;
    #pragma unroll
    for (int v = 0; v < 8; ++v) {
      const int m = half * 8 + v;
      MALL[m * 128 + p] = mAll[v];
      SALL[m * 128 + p] = sAll[v];
      MPOS[m * 128 + p] = mP[v];
      SPOS[m * 128 + p] = sP[v];
      SUMP[m * 128 + p] = suP[v];
      CNTP[m * 128 + p] = cnP[v];
      CAND[m * 256 + 2 * p]     = t1[v];
      CAND[m * 256 + 2 * p + 1] = t2[v];
    }
  }
  __syncthreads();

  // Phase 2: one thread per anchor row merges partials + selects top-K.
  if (tid < 16) {
    const int m = tid;
    float ma = NEGINF, sa = 0.f, mp_ = NEGINF, sp_ = 0.f, su = 0.f, cn = 0.f;
    for (int p = 0; p < 128; ++p) {
      lse_merge(ma,  sa,  MALL[m * 128 + p], SALL[m * 128 + p]);
      lse_merge(mp_, sp_, MPOS[m * 128 + p], SPOS[m * 128 + p]);
      su += SUMP[m * 128 + p];
      cn += CNTP[m * 128 + p];
    }

    const float* cd = CAND + m * 256;
    float hi = NEGINF;
    for (int i = 0; i < 256; ++i) hi = fmaxf(hi, cd[i]);

    int K = *topk;
    K = K < 1 ? 1 : (K > 256 ? 256 : K);
    const int negTotal = (B_ROWS - 1) - (int)cn;   // masks partition off-diag

    float lseNeg = NEGINF;
    if (negTotal > 0 && hi > NEGINF) {
      float lob = hi - 30.f, hib = hi;             // exp(<-30) is negligible
      int c0 = 0;
      for (int i = 0; i < 256; ++i) c0 += (cd[i] >= lob) ? 1 : 0;
      if (c0 > K) {
        for (int it = 0; it < 24; ++it) {          // bisect for K-th value
          float mid = 0.5f * (lob + hib);
          int cc = 0;
          for (int i = 0; i < 256; ++i) cc += (cd[i] >= mid) ? 1 : 0;
          if (cc >= K) lob = mid; else hib = mid;
        }
      }
      float s = 0.f;
      for (int i = 0; i < 256; ++i)
        if (cd[i] >= lob) s += __expf(cd[i] - hi);
      lseNeg = hi + __logf(s);
    }

    const float meanPos = su / fmaxf(cn, 1.f);
    const float lseAll  = ma + __logf(sa);
    const bool  vF = cn > 0.f;
    const float lsePos = vF ? (mp_ + __logf(sp_)) : NEGINF;
    const float md = fmaxf(lsePos, lseNeg);
    const float lseDen = (md > NEGINF)
        ? (md + __logf(__expf(lsePos - md) + __expf(lseNeg - md)))
        : NEGINF;
    const bool vM = vF && (negTotal > 0);

    const int rg = r0 + m;
    lossF[rg] = vF ? (lseAll - meanPos) : 0.f;
    valF[rg]  = vF ? 1.f : 0.f;
    lossM[rg] = vM ? (lseDen - meanPos) : 0.f;
    valM[rg]  = vM ? 1.f : 0.f;
  }
}

// ---------------------------------------------------------------- kernel 2 --
__global__ void __launch_bounds__(256)
supcon_final(const float* __restrict__ lossF, const float* __restrict__ valF,
             const float* __restrict__ lossM, const float* __restrict__ valM,
             const float* __restrict__ alphaPtr, float* __restrict__ out) {
  __shared__ float s0[256], s1[256], s2[256], s3[256];
  const int t = threadIdx.x;
  float a = 0.f, b = 0.f, c = 0.f, d = 0.f;
  for (int i = t; i < B_ROWS; i += 256) {
    a += lossF[i]; b += valF[i]; c += lossM[i]; d += valM[i];
  }
  s0[t] = a; s1[t] = b; s2[t] = c; s3[t] = d;
  __syncthreads();
  for (int off = 128; off > 0; off >>= 1) {
    if (t < off) {
      s0[t] += s0[t + off]; s1[t] += s1[t + off];
      s2[t] += s2[t + off]; s3[t] += s3[t + off];
    }
    __syncthreads();
  }
  if (t == 0) {
    const float nF = s1[0], nM = s3[0];
    const float lf = (nF > 0.f) ? (s0[0] / nF) : 0.f;
    const float lm = (nM > 0.f) ? (s2[0] / nM) : lf;
    const float al = *alphaPtr;
    out[0] = (1.f - al) * lf + al * lm;
  }
}

// ---------------------------------------------------------------------------
extern "C" void kernel_launch(void* const* d_in, const int* in_sizes, int n_in,
                              void* d_out, int out_size, void* d_ws,
                              size_t ws_size, hipStream_t stream) {
  const float* z      = (const float*)d_in[0];
  const int*   labels = (const int*)d_in[1];
  const int*   topk   = (const int*)d_in[2];
  const float* alpha  = (const float*)d_in[3];
  float* out = (float*)d_out;

  char* w = (char*)d_ws;
  unsigned short* zb = (unsigned short*)w;                 // 4 MB bf16 copy
  float* lossF = (float*)(w + (size_t)B_ROWS * DDIM * 2);  // per-row arrays
  float* valF  = lossF + B_ROWS;
  float* lossM = valF + B_ROWS;
  float* valM  = lossM + B_ROWS;

  supcon_cvt<<<(B_ROWS * DDIM) / (256 * 4), 256, 0, stream>>>(z, zb);
  supcon_main<<<B_ROWS / 16, 256, SMEM_BYTES, stream>>>(
      zb, labels, topk, lossF, valF, lossM, valM);
  supcon_final<<<1, 256, 0, stream>>>(lossF, valF, lossM, valM, alpha, out);
}
